// EncoderLayer_76347338654190
// MI455X (gfx1250) — compile-verified, matrix operand order checked by reading, not run on hardware
//
#include <hip/hip_runtime.h>

// ---------------------------------------------------------------------------
// Types for CDNA5 WMMA / TDM
// ---------------------------------------------------------------------------
typedef __bf16 bf16;
typedef __attribute__((ext_vector_type(16))) __bf16 v16bf;
typedef __attribute__((ext_vector_type(8)))  float  v8f;
typedef __attribute__((ext_vector_type(2)))  float  v2f;
typedef __attribute__((ext_vector_type(4)))  unsigned uint32x4;
typedef __attribute__((ext_vector_type(8)))  int      int32x8;
typedef __attribute__((ext_vector_type(4)))  int      int32x4;

#ifndef __has_builtin
#define __has_builtin(x) 0
#endif
#if __has_builtin(__builtin_amdgcn_tensor_load_to_lds) && __has_builtin(__builtin_amdgcn_s_wait_tensorcnt)
#define USE_TDM 1
#else
#define USE_TDM 0
#endif

// Problem constants (match the reference setup)
constexpr int NHn = 32, HDn = 64, MBn = 16;
constexpr int Bc = 4, Lc = 2048, Cc = 2048, DHc = 1024;
constexpr float ROPE_THETA = 10000.0f;

__device__ __forceinline__ bf16 f2bf(float f) {
  unsigned u = __builtin_bit_cast(unsigned, f);
  unsigned r = (u + 0x7FFFu + ((u >> 16) & 1u)) >> 16;
  unsigned short s = (unsigned short)r;
  return __builtin_bit_cast(bf16, s);
}

__device__ __forceinline__ unsigned lds_off32(const void* p) {
  // LDS aperture: flat_addr[31:0] == LDS byte offset
  return (unsigned)(size_t)p;
}

#if USE_TDM
// ---------------------------------------------------------------------------
// Tensor Data Mover: 2D tile (tile_x elements per row, tile_y rows) from
// row-major global (row stride = stride_elems) into LDS at lds_off, with
// LDS padding: (pad_amount+1) dwords inserted every (1<<(pad_interval+1)) dwords.
// D# layout per CDNA5 ISA 8.3/8.4 (group0 128b, group1 256b; groups 2/3 zero).
// 6-arg builtin: (uint32x4 g0, int32x8 g1, int32x4, int32x4, int32x8, i32 cpol)
// ---------------------------------------------------------------------------
__device__ __forceinline__ void tdm_load_2d(unsigned lds_off, const void* gptr,
                                            unsigned tile_x, unsigned tile_y,
                                            unsigned long long stride_elems,
                                            unsigned data_size_code,  // 0=1B 1=2B 2=4B 3=8B
                                            unsigned pad_interval_code,
                                            unsigned pad_amount_code) {
  unsigned long long ga = (unsigned long long)(size_t)gptr;
  uint32x4 g0;
  g0[0] = 1u;  // count=1, user descriptor
  g0[1] = lds_off;
  g0[2] = (unsigned)(ga & 0xFFFFFFFFull);
  g0[3] = (unsigned)((ga >> 32) & 0x01FFFFFFull) | 0x80000000u;  // type=2 ("image")
  const unsigned dim0 = 0x40000000u, dim1 = 0x40000000u;         // no OOB clipping
  int32x8 g1;
  g1[0] = (int)((data_size_code << 16) | (1u << 20) /*pad_enable*/ |
                (pad_interval_code << 22) | (pad_amount_code << 25));
  g1[1] = (int)((dim0 & 0xFFFFu) << 16);
  g1[2] = (int)(((dim0 >> 16) & 0xFFFFu) | ((dim1 & 0xFFFFu) << 16));
  g1[3] = (int)(((dim1 >> 16) & 0xFFFFu) | (tile_x << 16));
  g1[4] = (int)(tile_y & 0xFFFFu);
  g1[5] = (int)(stride_elems & 0xFFFFFFFFull);
  g1[6] = (int)((stride_elems >> 32) & 0xFFFFull);
  g1[7] = 0;
  int32x4 z4 = {0, 0, 0, 0};
  int32x8 z8 = {0, 0, 0, 0, 0, 0, 0, 0};
  __builtin_amdgcn_tensor_load_to_lds(g0, g1, z4, z4, z8, 0);
}
#endif

// ---------------------------------------------------------------------------
// f32 -> bf16 elementwise convert (weights)
// ---------------------------------------------------------------------------
__global__ void cvt_bf16_kernel(const float* __restrict__ x, bf16* __restrict__ y, size_t n) {
  size_t i = (size_t)blockIdx.x * 256 + threadIdx.x;
  size_t stride = (size_t)gridDim.x * 256;
  for (; i < n; i += stride) y[i] = f2bf(x[i]);
}

// ---------------------------------------------------------------------------
// LayerNorm over last dim D (256 threads per row). Optional f32 and bf16 outs.
// ---------------------------------------------------------------------------
__global__ __launch_bounds__(256) void ln_kernel(
    const float* __restrict__ x, const float* __restrict__ w, const float* __restrict__ b,
    float* __restrict__ y32, bf16* __restrict__ y16, int D, float eps) {
  size_t row = blockIdx.x;
  const float* xr = x + row * (size_t)D;
  int t = threadIdx.x;
  float s = 0.f, s2 = 0.f;
  for (int c = t; c < D; c += 256) { float v = xr[c]; s += v; s2 += v * v; }
  for (int o = 16; o > 0; o >>= 1) { s += __shfl_xor(s, o, 32); s2 += __shfl_xor(s2, o, 32); }
  __shared__ float ss[8], ss2[8];
  if ((t & 31) == 0) { ss[t >> 5] = s; ss2[t >> 5] = s2; }
  __syncthreads();
  float S = 0.f, S2 = 0.f;
#pragma unroll
  for (int i = 0; i < 8; i++) { S += ss[i]; S2 += ss2[i]; }
  float mu = S / D;
  float var = S2 / D - mu * mu;
  float rstd = rsqrtf(var + eps);
  for (int c = t; c < D; c += 256) {
    float v = (xr[c] - mu) * rstd * w[c] + b[c];
    if (y32) y32[row * (size_t)D + c] = v;
    if (y16) y16[row * (size_t)D + c] = f2bf(v);
  }
}

// ---------------------------------------------------------------------------
// bf16 x bf16 -> f32 GEMM, v_wmma_f32_16x16x32_bf16, TDM-staged double buffer.
//   Block tile 128x128, BK=32; 8 waves (2x4), each 64x32 (4x2 WMMA tiles).
//   LDS A: [m][k] stride 40 bf16 (TDM pad 16+4 dwords)
//   LDS B: [k][n] stride 130 bf16 (TDM pad 64+1 dwords)
// EPI: 0 plain f32; 1 +residual f32; 2 +bias+ReLU bf16; 3 +bias+residual f32
// ---------------------------------------------------------------------------
template <int EPI>
__global__ __launch_bounds__(256) void gemm_bf16_kernel(
    const bf16* __restrict__ A, const bf16* __restrict__ Bm,
    float* __restrict__ Cf, bf16* __restrict__ Cb,
    const float* __restrict__ bias, const float* __restrict__ res,
    int M, int N, int K) {
  constexpr int BM = 128, BN = 128, BK = 32;
  constexpr int LDA = 40;    // bf16 per A row
  constexpr int LDBk = 130;  // bf16 per B k-row
  __shared__ bf16 sA[2][BM * LDA];
  __shared__ bf16 sB[2][BK * LDBk];

  int t = threadIdx.x;
  int lane = t & 31, w = t >> 5;
  int wr = w >> 2, wc = w & 3;
  int bm = blockIdx.y * BM, bn = blockIdx.x * BN;
  int hi = lane >> 4, lo = lane & 15;

  v8f acc[4][2];
#pragma unroll
  for (int mt = 0; mt < 4; mt++)
#pragma unroll
    for (int nt = 0; nt < 2; nt++)
#pragma unroll
      for (int v = 0; v < 8; v++) acc[mt][nt][v] = 0.f;

#if USE_TDM
  if (w == 0) {
    tdm_load_2d(lds_off32(&sA[0][0]), A + (size_t)bm * K, BK, BM, (unsigned long long)K, 1, 3, 3);
    tdm_load_2d(lds_off32(&sB[0][0]), Bm + bn, BN, BK, (unsigned long long)N, 1, 5, 0);
  }
#endif

  for (int kb = 0; kb < K; kb += BK) {
#if USE_TDM
    int buf = (kb >> 5) & 1;
    if (w == 0) {
      if (kb + BK < K) {
        tdm_load_2d(lds_off32(&sA[buf ^ 1][0]), A + (size_t)bm * K + kb + BK, BK, BM,
                    (unsigned long long)K, 1, 3, 3);
        tdm_load_2d(lds_off32(&sB[buf ^ 1][0]), Bm + (size_t)(kb + BK) * N + bn, BN, BK,
                    (unsigned long long)N, 1, 5, 0);
        __builtin_amdgcn_s_wait_tensorcnt(2);
      } else {
        __builtin_amdgcn_s_wait_tensorcnt(0);
      }
    }
    __syncthreads();
#else
    int buf = 0;
    if (kb + BK < K) {
      __builtin_prefetch(A + (size_t)(bm + (t >> 1)) * K + kb + BK, 0, 1);
      __builtin_prefetch(Bm + (size_t)(kb + BK + (t >> 3)) * N + bn + (t & 7) * 16, 0, 1);
    }
    {  // A tile
      int m = t >> 1, half = t & 1;
      const uint4* s4 = (const uint4*)(A + (size_t)(bm + m) * K + kb + half * 16);
      uint4 a0 = s4[0], a1 = s4[1];
      uint4* d4 = (uint4*)&sA[0][m * LDA + half * 16];
      d4[0] = a0; d4[1] = a1;
    }
    {  // B tile ([k][n], dword stores)
      int k = t >> 3, seg = t & 7;
      const unsigned* s32 = (const unsigned*)(Bm + (size_t)(kb + k) * N + bn + seg * 16);
      unsigned* d32 = (unsigned*)&sB[0][k * LDBk + seg * 16];
#pragma unroll
      for (int e = 0; e < 8; e++) d32[e] = s32[e];
    }
    __syncthreads();
#endif

    const bf16* sAb = &sA[buf][0];
    const unsigned short* sBu = (const unsigned short*)&sB[buf][0];
    union U { v16bf v; unsigned u[8]; };
    v16bf afr[4], bfr[2];
#pragma unroll
    for (int mt = 0; mt < 4; mt++) {
      U u; int m = wr * 64 + mt * 16 + lo;
#pragma unroll
      for (int v = 0; v < 8; v++) {
        int kk = (v >> 2) * 16 + hi * 8 + (v & 3) * 2;  // ISA 16-bit A 16x32 layout
        u.u[v] = *(const unsigned*)&sAb[m * LDA + kk];
      }
      afr[mt] = u.v;
    }
#pragma unroll
    for (int nt = 0; nt < 2; nt++) {
      U u; int n = wc * 32 + nt * 16 + lo;
#pragma unroll
      for (int v = 0; v < 8; v++) {
        int kk = (v >> 2) * 16 + hi * 8 + (v & 3) * 2;
        unsigned x0 = sBu[kk * LDBk + n];
        unsigned x1 = sBu[(kk + 1) * LDBk + n];
        u.u[v] = x0 | (x1 << 16);
      }
      bfr[nt] = u.v;
    }
#pragma unroll
    for (int mt = 0; mt < 4; mt++)
#pragma unroll
      for (int nt = 0; nt < 2; nt++)
        acc[mt][nt] = __builtin_amdgcn_wmma_f32_16x16x32_bf16(
            false, afr[mt], false, bfr[nt], (short)0, acc[mt][nt], false, false);
    __syncthreads();
  }

#pragma unroll
  for (int mt = 0; mt < 4; mt++)
#pragma unroll
    for (int nt = 0; nt < 2; nt++) {
      int col = bn + wc * 32 + nt * 16 + lo;
#pragma unroll
      for (int v = 0; v < 8; v++) {
        int row = bm + wr * 64 + mt * 16 + v + 8 * hi;
        size_t idx = (size_t)row * N + col;
        float val = acc[mt][nt][v];
        if constexpr (EPI == 0) {
          Cf[idx] = val;
        } else if constexpr (EPI == 1) {
          Cf[idx] = val + res[idx];
        } else if constexpr (EPI == 2) {
          val += bias[col];
          Cb[idx] = f2bf(fmaxf(val, 0.f));
        } else {
          Cf[idx] = val + bias[col] + res[idx];
        }
      }
    }
}

// ---------------------------------------------------------------------------
// RoPE on XQ, XK in (B,L,C) layout; pos = l % MB, pairs (j, j+32) per head.
// ---------------------------------------------------------------------------
__global__ void rope_kernel(float* __restrict__ XQ, float* __restrict__ XK) {
  size_t total = (size_t)Bc * Lc * NHn * 32;
  size_t i = (size_t)blockIdx.x * 256 + threadIdx.x;
  if (i >= total) return;
  int j = (int)(i & 31);
  size_t r = i >> 5;
  int h = (int)(r % NHn);
  size_t bl = r / NHn;
  int l = (int)(bl % Lc);
  float pos = (float)(l & (MBn - 1));
  float inv = __expf(-(2.f * j / (float)HDn) * __logf(ROPE_THETA));
  float ang = pos * inv;
  float s, c;
  __sincosf(ang, &s, &c);
  size_t base = bl * Cc + (size_t)h * HDn;
  float q1 = XQ[base + j], q2 = XQ[base + j + 32];
  XQ[base + j]      = q1 * c - q2 * s;
  XQ[base + j + 32] = q2 * c + q1 * s;
  float k1 = XK[base + j], k2 = XK[base + j + 32];
  XK[base + j]      = k1 * c - k2 * s;
  XK[base + j + 32] = k2 * c + k1 * s;
}

// ---------------------------------------------------------------------------
// Per-token TTT learning rate: lr[b,h,l] = sigmoid(x_ln[b,l,:]·ttt_lr_w[h,:]+b[h])
// ---------------------------------------------------------------------------
__global__ __launch_bounds__(256) void lr_kernel(
    const float* __restrict__ xln, const float* __restrict__ lr_w,
    const float* __restrict__ lr_b, float* __restrict__ lr_out) {
  size_t row = blockIdx.x;  // b*L + l
  __shared__ float xr[Cc];
  int t = threadIdx.x;
  for (int c = t; c < Cc; c += 256) xr[c] = xln[row * Cc + c];
  __syncthreads();
  int h = t >> 3, p = t & 7;
  const float* wrow = lr_w + (size_t)h * Cc;
  float s = 0.f;
  int c0 = p * (Cc / 8);
  for (int c = c0; c < c0 + Cc / 8; c++) s += xr[c] * wrow[c];
  for (int o = 4; o > 0; o >>= 1) s += __shfl_down(s, o, 32);
  if (p == 0) {
    float v = 1.f / (1.f + __expf(-(s + lr_b[h])));
    size_t b = row / Lc;
    int l = (int)(row % Lc);
    lr_out[(b * NHn + h) * (size_t)Lc + l] = v;
  }
}

// ---------------------------------------------------------------------------
// TTT sequential scan. One wave per (b,h).
// State kept TRANSPOSED in LDS: WT[j][k] = W1[k][j]  (so all W-operand
// B-fragments are contiguous ds_load_b64). Gradient kept transposed too:
// gradT[j][i]. All matmuls via v_wmma_f32_16x16x4_f32; tiles staged by TDM.
// ---------------------------------------------------------------------------
__device__ __forceinline__ v8f wmma_f32(v2f a, v2f b, v8f c) {
  return __builtin_amdgcn_wmma_f32_16x16x4_f32(false, a, false, b, (short)0, c, false, false);
}

__global__ __launch_bounds__(32) void ttt_scan_kernel(
    const float* __restrict__ XQ, const float* __restrict__ XK, const float* __restrict__ XV,
    const float* __restrict__ lr_all, const float* __restrict__ tok_off,
    const float* __restrict__ W1_0, const float* __restrict__ b1_0,
    const float* __restrict__ lnw, const float* __restrict__ lnb,
    float* __restrict__ Yout) {
  constexpr int S = 68;   // padded f32 stride (64 + 4 dwords, matches TDM pad)
  constexpr int SG = 20;  // gradT / EA row stride
  __shared__ float WT[64 * S];  // WT[j*S + k] = W1[k][j]
  __shared__ float b1v[64], g[64], bb[64], lrs[16], tokv[16];
  __shared__ float xqb[2][16 * S], xkb[2][16 * S], xvb[2][16 * S];
  __shared__ float Z1[16 * S], zb[16 * S];
  __shared__ float gradT[64 * SG];  // gradT[j*SG + i] = grad[i][j]
  __shared__ float EA[16 * SG];

  int bh = blockIdx.x;
  int b = bh / NHn, h = bh % NHn;
  int lane = threadIdx.x;
  int lo = lane & 15, hi = lane >> 4;

  for (int i = lane; i < 64 * 64; i += 32) {
    int k = i >> 6, j = i & 63;
    WT[j * S + k] = W1_0[(size_t)h * 4096 + i];  // transpose on load
  }
  for (int j = lane; j < 64; j += 32) {
    b1v[j] = b1_0[h * 64 + j];
    g[j] = lnw[h * 64 + j];
    bb[j] = lnb[h * 64 + j];
  }
  if (lane < 16) tokv[lane] = fmaxf(1.f / (lane + 1) + tok_off[lane], 0.f);
  __syncthreads();

#if USE_TDM
  {
    size_t g0 = (size_t)b * Lc * Cc + (size_t)h * 64;
    tdm_load_2d(lds_off32(&xqb[0][0]), XQ + g0, 64, 16, (unsigned long long)Cc, 2, 5, 3);
    tdm_load_2d(lds_off32(&xkb[0][0]), XK + g0, 64, 16, (unsigned long long)Cc, 2, 5, 3);
    tdm_load_2d(lds_off32(&xvb[0][0]), XV + g0, 64, 16, (unsigned long long)Cc, 2, 5, 3);
  }
#endif

  for (int n = 0; n < Lc / MBn; n++) {
    size_t rowbase = (size_t)b * Lc + (size_t)n * 16;
#if USE_TDM
    int buf = n & 1;
    if (n + 1 < Lc / MBn) {
      size_t gn = (rowbase + 16) * Cc + (size_t)h * 64;
      tdm_load_2d(lds_off32(&xqb[buf ^ 1][0]), XQ + gn, 64, 16, (unsigned long long)Cc, 2, 5, 3);
      tdm_load_2d(lds_off32(&xkb[buf ^ 1][0]), XK + gn, 64, 16, (unsigned long long)Cc, 2, 5, 3);
      tdm_load_2d(lds_off32(&xvb[buf ^ 1][0]), XV + gn, 64, 16, (unsigned long long)Cc, 2, 5, 3);
      __builtin_amdgcn_s_wait_tensorcnt(3);
    } else {
      __builtin_amdgcn_s_wait_tensorcnt(0);
    }
    const float* xq = xqb[buf];
    const float* xk = xkb[buf];
    const float* xv = xvb[buf];
#else
    for (int t = lane; t < 16 * 16; t += 32) {
      int r = t >> 4, c4 = (t & 15) * 4;
      size_t gidx = (rowbase + r) * Cc + (size_t)h * 64 + c4;
      *(float4*)&xqb[0][r * S + c4] = *(const float4*)&XQ[gidx];
      *(float4*)&xkb[0][r * S + c4] = *(const float4*)&XK[gidx];
      *(float4*)&xvb[0][r * S + c4] = *(const float4*)&XV[gidx];
    }
    const float* xq = xqb[0];
    const float* xk = xkb[0];
    const float* xv = xvb[0];
#endif
    if (lane < 16) lrs[lane] = lr_all[((size_t)b * NHn + h) * Lc + n * 16 + lane];
    __syncthreads();

    // ---- Z1 = xk @ W1 + b1 (16x64, K=64); A-frags hoisted, all ops b64 ----
    {
      v2f ak[16];
#pragma unroll
      for (int ks = 0; ks < 16; ks++) ak[ks] = *(const v2f*)&xk[lo * S + ks * 4 + hi * 2];
#pragma unroll
      for (int nt = 0; nt < 4; nt++) {
        v8f c;
        float bv = b1v[nt * 16 + lo];
#pragma unroll
        for (int v = 0; v < 8; v++) c[v] = bv;
#pragma unroll
        for (int ks = 0; ks < 16; ks++) {
          int k0 = ks * 4 + hi * 2;
          v2f bf = *(const v2f*)&WT[(nt * 16 + lo) * S + k0];  // contiguous
          c = wmma_f32(ak[ks], bf, c);
        }
#pragma unroll
        for (int v = 0; v < 8; v++) Z1[(v + 8 * hi) * S + nt * 16 + lo] = c[v];
      }
    }
    __syncthreads();

    // ---- gradT = ln_l2_bwd(Z1, xv - xk, g, bb)^T : 2 lanes per row ----
    {
      int r = lo, half = hi;
      float s1 = 0.f, s2 = 0.f;
      for (int c = half * 32; c < half * 32 + 32; c++) { float z = Z1[r * S + c]; s1 += z; s2 += z * z; }
      s1 += __shfl_xor(s1, 16, 32);
      s2 += __shfl_xor(s2, 16, 32);
      float mu = s1 * (1.f / 64.f);
      float var = s2 * (1.f / 64.f) - mu * mu;
      float rstd = rsqrtf(var + 1e-6f);
      float sg = 0.f, sgx = 0.f;
      for (int c = half * 32; c < half * 32 + 32; c++) {
        float xh = (Z1[r * S + c] - mu) * rstd;
        float tgt = xv[r * S + c] - xk[r * S + c];
        float gx = (g[c] * xh + bb[c] - tgt) * g[c];
        zb[r * S + c] = gx;  // temp
        sg += gx; sgx += gx * xh;
      }
      sg += __shfl_xor(sg, 16, 32);
      sgx += __shfl_xor(sgx, 16, 32);
      for (int c = half * 32; c < half * 32 + 32; c++) {
        float xh = (Z1[r * S + c] - mu) * rstd;
        float gx = zb[r * S + c];
        gradT[c * SG + r] = (64.f * gx - sg - xh * sgx) * (rstd * (1.f / 64.f));
      }
    }
    __syncthreads();

    // ---- Attn = tril(xq @ xk^T); EA = et .* (Attn + tril) ----
    {
      v8f c;
#pragma unroll
      for (int v = 0; v < 8; v++) c[v] = 0.f;
#pragma unroll
      for (int ks = 0; ks < 16; ks++) {
        int k0 = ks * 4 + hi * 2;
        v2f a = *(const v2f*)&xq[lo * S + k0];
        v2f bf = *(const v2f*)&xk[lo * S + k0];  // B = xk^T
        c = wmma_f32(a, bf, c);
      }
      int p = lo;
#pragma unroll
      for (int v = 0; v < 8; v++) {
        int i = v + 8 * hi;
        float a1 = (p <= i) ? (c[v] + 1.f) : 0.f;
        EA[i * SG + p] = tokv[i] * lrs[p] * (1.f / 64.f) * a1;
      }
    }
    __syncthreads();

    // ---- Z1bar = xq @ W1 + b1 - EA @ grad  (all contiguous operands) ----
    {
      v2f aq[16], ae[4];
#pragma unroll
      for (int ks = 0; ks < 16; ks++) aq[ks] = *(const v2f*)&xq[lo * S + ks * 4 + hi * 2];
#pragma unroll
      for (int ks = 0; ks < 4; ks++) ae[ks] = -(*(const v2f*)&EA[lo * SG + ks * 4 + hi * 2]);
#pragma unroll
      for (int nt = 0; nt < 4; nt++) {
        v8f c;
        float bv = b1v[nt * 16 + lo];
#pragma unroll
        for (int v = 0; v < 8; v++) c[v] = bv;
#pragma unroll
        for (int ks = 0; ks < 16; ks++) {
          int k0 = ks * 4 + hi * 2;
          v2f bf = *(const v2f*)&WT[(nt * 16 + lo) * S + k0];
          c = wmma_f32(aq[ks], bf, c);
        }
#pragma unroll
        for (int ks = 0; ks < 4; ks++) {
          int k0 = ks * 4 + hi * 2;
          v2f bf = *(const v2f*)&gradT[(nt * 16 + lo) * SG + k0];
          c = wmma_f32(ae[ks], bf, c);
        }
#pragma unroll
        for (int v = 0; v < 8; v++) zb[(v + 8 * hi) * S + nt * 16 + lo] = c[v];
      }
    }
    __syncthreads();

    // ---- out = xq + LN(Z1bar) -> global ----
    {
      int r = lo, half = hi;
      float s1 = 0.f, s2 = 0.f;
      for (int c = half * 32; c < half * 32 + 32; c++) { float z = zb[r * S + c]; s1 += z; s2 += z * z; }
      s1 += __shfl_xor(s1, 16, 32);
      s2 += __shfl_xor(s2, 16, 32);
      float mu = s1 * (1.f / 64.f);
      float var = s2 * (1.f / 64.f) - mu * mu;
      float rstd = rsqrtf(var + 1e-6f);
      for (int c = half * 32; c < half * 32 + 32; c++) {
        float val = xq[r * S + c] + ((zb[r * S + c] - mu) * rstd * g[c] + bb[c]);
        Yout[(rowbase + r) * Cc + (size_t)h * 64 + c] = val;
      }
    }
    __syncthreads();

    // ---- WT[j][k] -= sum_i le[i]*grad[i][j]*xk[i][k]  (64x64, K=16) ----
    {
      float le_s = tokv[15] * (1.f / 64.f);
      v2f le2[4];
#pragma unroll
      for (int ks = 0; ks < 4; ks++) {
        int i0 = ks * 4 + hi * 2;
        le2[ks].x = -le_s * lrs[i0];
        le2[ks].y = -le_s * lrs[i0 + 1];
      }
#pragma unroll
      for (int jt = 0; jt < 4; jt++) {
        v2f afr2[4];
        int j = jt * 16 + lo;
#pragma unroll
        for (int ks = 0; ks < 4; ks++) {
          int i0 = ks * 4 + hi * 2;
          v2f gpair = *(const v2f*)&gradT[j * SG + i0];  // contiguous
          afr2[ks] = gpair * le2[ks];
        }
#pragma unroll
        for (int kt = 0; kt < 4; kt++) {
          v8f c;
#pragma unroll
          for (int v = 0; v < 8; v++) c[v] = WT[(jt * 16 + v + 8 * hi) * S + kt * 16 + lo];
#pragma unroll
          for (int ks = 0; ks < 4; ks++) {
            int i0 = ks * 4 + hi * 2;
            v2f bf;
            bf.x = xk[i0 * S + kt * 16 + lo];
            bf.y = xk[(i0 + 1) * S + kt * 16 + lo];
            c = wmma_f32(afr2[ks], bf, c);
          }
#pragma unroll
          for (int v = 0; v < 8; v++) WT[(jt * 16 + v + 8 * hi) * S + kt * 16 + lo] = c[v];
        }
      }
    }
    // ---- b1 -= sum_i last_eta[i] * grad[i,:]  (contiguous gradT rows) ----
    {
      float le_s = tokv[15] * (1.f / 64.f);
      for (int j = lane; j < 64; j += 32) {
        float s = 0.f;
#pragma unroll
        for (int i = 0; i < 16; i++) s += lrs[i] * gradT[j * SG + i];
        b1v[j] -= le_s * s;
      }
    }
    __syncthreads();
  }
}

// ---------------------------------------------------------------------------
// Host-side launch
// ---------------------------------------------------------------------------
extern "C" void kernel_launch(void* const* d_in, const int* in_sizes, int n_in,
                              void* d_out, int out_size, void* d_ws, size_t ws_size,
                              hipStream_t stream) {
  (void)in_sizes; (void)n_in; (void)out_size; (void)ws_size;
  const float* enc     = (const float*)d_in[0];
  const float* ln1_w   = (const float*)d_in[1];
  const float* ln1_b   = (const float*)d_in[2];
  const float* q_w     = (const float*)d_in[3];
  const float* k_w     = (const float*)d_in[4];
  const float* v_w     = (const float*)d_in[5];
  const float* o_w     = (const float*)d_in[6];
  const float* lr_w    = (const float*)d_in[7];
  const float* lr_b    = (const float*)d_in[8];
  const float* tok_off = (const float*)d_in[9];
  const float* tnw     = (const float*)d_in[10];
  const float* tnb     = (const float*)d_in[11];
  const float* W1_0    = (const float*)d_in[12];
  const float* b1_0    = (const float*)d_in[13];
  const float* post_w  = (const float*)d_in[14];
  const float* post_b  = (const float*)d_in[15];
  const float* ffn_w   = (const float*)d_in[16];
  const float* ffn_b   = (const float*)d_in[17];
  const float* w1f     = (const float*)d_in[18];
  const float* b1f     = (const float*)d_in[19];
  const float* w2f     = (const float*)d_in[20];
  const float* b2f     = (const float*)d_in[21];
  float* out = (float*)d_out;

  const size_t BL  = (size_t)Bc * Lc;
  const size_t BLC = BL * Cc;

  char* ws = (char*)d_ws;
  size_t off = 0;
  auto alloc = [&](size_t bytes) -> void* {
    void* p = ws + off;
    off += (bytes + 255) & ~(size_t)255;
    return p;
  };
  float* xln32 = (float*)alloc(BLC * 4);
  bf16* xln16  = (bf16*)alloc(BLC * 2);
  bf16* wq16   = (bf16*)alloc((size_t)Cc * Cc * 2);
  bf16* wk16   = (bf16*)alloc((size_t)Cc * Cc * 2);
  bf16* wv16   = (bf16*)alloc((size_t)Cc * Cc * 2);
  bf16* wo16   = (bf16*)alloc((size_t)Cc * Cc * 2);
  bf16* w1f16  = (bf16*)alloc((size_t)Cc * DHc * 2);
  bf16* w2f16  = (bf16*)alloc((size_t)DHc * Cc * 2);
  float* XQ    = (float*)alloc(BLC * 4);
  float* XK    = (float*)alloc(BLC * 4);
  float* XV    = (float*)alloc(BLC * 4);
  float* lrbuf = (float*)alloc((size_t)Bc * NHn * Lc * 4);
  float* ttt_y = (float*)alloc(BLC * 4);
  bf16* ypost16 = (bf16*)alloc(BLC * 2);
  float* yres  = (float*)alloc(BLC * 4);
  bf16* zln16  = (bf16*)alloc(BLC * 2);
  bf16* ffnh16 = (bf16*)alloc(BL * DHc * 2);

  {
    size_t nw = (size_t)Cc * Cc;
    int gsz = 4096;
    cvt_bf16_kernel<<<gsz, 256, 0, stream>>>(q_w, wq16, nw);
    cvt_bf16_kernel<<<gsz, 256, 0, stream>>>(k_w, wk16, nw);
    cvt_bf16_kernel<<<gsz, 256, 0, stream>>>(v_w, wv16, nw);
    cvt_bf16_kernel<<<gsz, 256, 0, stream>>>(o_w, wo16, nw);
    cvt_bf16_kernel<<<gsz, 256, 0, stream>>>(w1f, w1f16, (size_t)Cc * DHc);
    cvt_bf16_kernel<<<gsz, 256, 0, stream>>>(w2f, w2f16, (size_t)DHc * Cc);
  }
  ln_kernel<<<(int)BL, 256, 0, stream>>>(enc, ln1_w, ln1_b, xln32, xln16, Cc, 1e-5f);
  {
    dim3 grid(Cc / 128, (int)(BL / 128));
    gemm_bf16_kernel<0><<<grid, 256, 0, stream>>>(xln16, wq16, XQ, nullptr, nullptr, nullptr, (int)BL, Cc, Cc);
    gemm_bf16_kernel<0><<<grid, 256, 0, stream>>>(xln16, wk16, XK, nullptr, nullptr, nullptr, (int)BL, Cc, Cc);
    gemm_bf16_kernel<0><<<grid, 256, 0, stream>>>(xln16, wv16, XV, nullptr, nullptr, nullptr, (int)BL, Cc, Cc);
  }
  {
    size_t total = BL * NHn * 32;
    rope_kernel<<<(int)((total + 255) / 256), 256, 0, stream>>>(XQ, XK);
  }
  lr_kernel<<<(int)BL, 256, 0, stream>>>(xln32, lr_w, lr_b, lrbuf);
  ttt_scan_kernel<<<Bc * NHn, 32, 0, stream>>>(XQ, XK, XV, lrbuf, tok_off, W1_0, b1_0, tnw, tnb, ttt_y);
  ln_kernel<<<(int)BL, 256, 0, stream>>>(ttt_y, post_w, post_b, nullptr, ypost16, Cc, 1e-6f);
  {
    dim3 grid(Cc / 128, (int)(BL / 128));
    gemm_bf16_kernel<1><<<grid, 256, 0, stream>>>(ypost16, wo16, yres, nullptr, nullptr, enc, (int)BL, Cc, Cc);
  }
  ln_kernel<<<(int)BL, 256, 0, stream>>>(yres, ffn_w, ffn_b, nullptr, zln16, Cc, 1e-6f);
  {
    dim3 grid(DHc / 128, (int)(BL / 128));
    gemm_bf16_kernel<2><<<grid, 256, 0, stream>>>(zln16, w1f16, nullptr, ffnh16, b1f, nullptr, (int)BL, DHc, Cc);
  }
  {
    dim3 grid(Cc / 128, (int)(BL / 128));
    gemm_bf16_kernel<3><<<grid, 256, 0, stream>>>(ffnh16, w2f16, out, nullptr, b2f, yres, (int)BL, Cc, DHc);
  }
}
